// CausalAttNet_90606630077047
// MI455X (gfx1250) — matrix operand order repaired and checked by reading, not compiled, not run on hardware
//
#include <hip/hip_runtime.h>
#include <hip/hip_bf16.h>
#include <stdint.h>

// Problem constants (match reference)
#define G    256
#define NPG  128
#define EPG  512
#define H    512
#define H2   1024
#define NN   (G*NPG)          // 32768 nodes
#define NE   (G*EPG)          // 131072 edges
#define NRES 256
#define GC   (G*NRES)         // 65536
#define NH   ((size_t)NN*H)   // 16777216
#define NH2  ((size_t)NN*H2)  // 33554432
#define EPS_BN 1e-5f

// Output layout (flat float, reference tuple return order)
#define O_X0   ((size_t)0)
#define O_EIC  (O_X0 + NH)
#define O_EAC  (O_EIC + 2*(size_t)GC)
#define O_PRC  (O_EAC + (size_t)GC)
#define O_B0   (O_PRC + (size_t)GC)
#define O_X1   (O_B0 + (size_t)NN)
#define O_EIF  (O_X1 + NH)
#define O_EAF  (O_EIF + 2*(size_t)GC)
#define O_PRF  (O_EAF + (size_t)GC)
#define O_B1   (O_PRF + (size_t)GC)
#define O_PRED (O_B1 + (size_t)NN)

typedef __attribute__((ext_vector_type(16))) __bf16 bf16x16;
typedef __attribute__((ext_vector_type(8)))  float  f32x8;

__device__ __forceinline__ uint16_t f2bf(float f){
  union{float f; uint32_t u;} x; x.f=f;
  uint32_t u = x.u;
  uint32_t r = u + 0x7FFFu + ((u>>16)&1u);   // round-to-nearest-even
  return (uint16_t)(r>>16);
}

// ---------------- element-wise / gather kernels ----------------

__global__ void embed_k(const int* __restrict__ xfeat,
                        const float* __restrict__ atom_emb,
                        const float* __restrict__ vn_emb,
                        float* __restrict__ out){
  size_t i = (size_t)blockIdx.x*256 + threadIdx.x;   // NH threads
  int n = (int)(i / H), j = (int)(i % H);
  out[i] = atom_emb[(size_t)xfeat[n]*H + j] + vn_emb[j];
}

__global__ void edge_msg_k(const float* __restrict__ h,
                           const float* __restrict__ bond,
                           const int* __restrict__ ei,
                           const int* __restrict__ ea,
                           float* __restrict__ agg){
  size_t t = (size_t)blockIdx.x*256 + threadIdx.x;   // NE*(H/4) threads
  int e = (int)(t / (H/4));
  int c = (int)(t % (H/4)) * 4;
  int src = ei[e], dst = ei[NE + e], a = ea[e];
  float4 hv = *(const float4*)(h    + (size_t)src*H + c);
  float4 bv = *(const float4*)(bond + (size_t)a  *H + c);
  float4 m;
  m.x = fmaxf(hv.x + bv.x, 0.f);
  m.y = fmaxf(hv.y + bv.y, 0.f);
  m.z = fmaxf(hv.z + bv.z, 0.f);
  m.w = fmaxf(hv.w + bv.w, 0.f);
  float* d = agg + (size_t)dst*H + c;
  atomicAdd(d+0, m.x); atomicAdd(d+1, m.y);
  atomicAdd(d+2, m.z); atomicAdd(d+3, m.w);
}

__global__ void combine_k(const float* __restrict__ h,
                          const float* __restrict__ agg,
                          const float* __restrict__ eps,
                          uint16_t* __restrict__ zb){
  size_t i = (size_t)blockIdx.x*256 + threadIdx.x;   // NH threads
  zb[i] = f2bf((1.f + eps[0])*h[i] + agg[i]);
}

__global__ void wt_k(const float* __restrict__ W, uint16_t* __restrict__ Wt,
                     int K, int Nc){
  int i = blockIdx.x*256 + threadIdx.x;              // K*Nc threads
  int k = i / Nc, n = i % Nc;
  Wt[(size_t)n*K + k] = f2bf(W[i]);                  // transpose to Nc x K
}

__global__ void tobf_k(const float* __restrict__ x, uint16_t* __restrict__ y){
  size_t i = (size_t)blockIdx.x*256 + threadIdx.x;
  y[i] = f2bf(x[i]);
}

__global__ void add_vn_k(const float* __restrict__ x,
                         const float* __restrict__ vnrow,
                         float* __restrict__ out){
  size_t i = (size_t)blockIdx.x*256 + threadIdx.x;   // NH threads
  int node = (int)(i / H), j = (int)(i % H);
  int g = node / NPG;
  out[i] = x[i] + vnrow[(size_t)g*H + j];
}

__global__ void segsum_vn_k(const float* __restrict__ hin,
                            const float* __restrict__ vn_emb,
                            float* __restrict__ vt){
  int tid = blockIdx.x*256 + threadIdx.x;            // G*H threads
  int g = tid / H, j = tid % H;
  float s = vn_emb[j];
  const float* p = hin + (size_t)g*NPG*H + j;
  #pragma unroll 4
  for (int n = 0; n < NPG; ++n) s += p[(size_t)n*H];
  vt[tid] = s;
}

// ---------------- batch-norm ----------------

__global__ void bn_stats_k(const float* __restrict__ X, int Mrows, int Ncols,
                           float* __restrict__ mean, float* __restrict__ rstd){
  int j = blockIdx.x;                                // one block per column
  __shared__ float s1[256], s2[256];
  float a = 0.f, b = 0.f;
  for (int i = threadIdx.x; i < Mrows; i += 256){
    float v = X[(size_t)i*Ncols + j];
    a += v; b += v*v;
  }
  s1[threadIdx.x] = a; s2[threadIdx.x] = b; __syncthreads();
  for (int s = 128; s > 0; s >>= 1){
    if (threadIdx.x < s){
      s1[threadIdx.x] += s1[threadIdx.x + s];
      s2[threadIdx.x] += s2[threadIdx.x + s];
    }
    __syncthreads();
  }
  if (threadIdx.x == 0){
    float m = s1[0] / (float)Mrows;
    float v = s2[0] / (float)Mrows - m*m;
    mean[j] = m;
    rstd[j] = rsqrtf(v + EPS_BN);
  }
}

__global__ void bn_apply_k(const float* __restrict__ X,
                           const float* __restrict__ mean,
                           const float* __restrict__ rstd,
                           const float* __restrict__ g,
                           const float* __restrict__ b,
                           const float* __restrict__ addsrc,
                           float* __restrict__ out,
                           int Ncols, int do_relu){
  size_t i = (size_t)blockIdx.x*256 + threadIdx.x;   // Mrows*Ncols threads
  int j = (int)(i % Ncols);
  float v = (X[i] - mean[j]) * rstd[j] * g[j] + b[j];
  if (do_relu) v = fmaxf(v, 0.f);
  if (addsrc) v += addsrc[i];
  out[i] = v;
}

// ---------------- WMMA bf16 GEMM: C[M x Nc] = A[M x K] * Bt^T + bias ----------------
// A: row-major M x K (bf16 bits). Bt: Nc x K (bf16 bits, weight pre-transposed).
// One wave computes a 32x64 output tile: per K-step it loads 2 A fragments and
// 4 B fragments and issues 8x v_wmma_f32_16x16x32_bf16 (B reused across the two
// M sub-tiles -> ~22 FLOP/B from cache, keeps the matrix pipe fed).
// Fragment layouts per CDNA5 ISA 7.12.2 (wave32).

__global__ __launch_bounds__(256) void gemm_bf16_wmma_k(
    const uint16_t* __restrict__ A,
    const uint16_t* __restrict__ Bt,
    const float* __restrict__ bias,
    float* __restrict__ C,
    int M, int Nc, int K)
{
  const int lane  = threadIdx.x & 31;
  const int wid   = threadIdx.x >> 5;
  const int waveM = wid >> 2;            // 0..1 -> 32 rows each
  const int waveN = wid & 3;             // 0..3 -> 64 cols each
  const int tileM = blockIdx.x*64 + waveM*32;
  const int tileN = blockIdx.y*256 + waveN*64;
  if (tileM >= M) return;

  const int rowA0 = tileM + (lane & 15);       // M sub-tile 0
  const int rowA1 = rowA0 + 16;                // M sub-tile 1
  const int kselA = (lane >> 4) * 8;           // A frag: lanes 16-31 hold K+8 group
  const int colB0 = (lane & 15);
  const int kselB = (lane >> 4) * 8;           // B frag: lanes 16-31 hold K+8 group

  f32x8 acc[2][4] = {};

  for (int kk = 0; kk < K; kk += 32){
    // A fragments: elements 0-7 -> K = kk+kselA+0..7 ; elements 8-15 -> +16
    union { uint4 u[2]; bf16x16 v; } a0, a1;
    {
      const uint4* ap = (const uint4*)(A + (size_t)rowA0*K + kk + kselA);
      a0.u[0] = ap[0];
      a0.u[1] = ap[2];                   // +16 bf16 elements
      __builtin_prefetch(ap + 8, 0, 3);  // K-block two steps ahead (+64 bf16) of this row, keep near
    }
    {
      const uint4* ap = (const uint4*)(A + (size_t)rowA1*K + kk + kselA);
      a1.u[0] = ap[0];
      a1.u[1] = ap[2];
      __builtin_prefetch(ap + 8, 0, 3);
    }

    #pragma unroll
    for (int t = 0; t < 4; ++t){
      union { uint4 u[2]; bf16x16 v; } bf_;
      const uint4* bp = (const uint4*)(Bt + (size_t)(tileN + t*16 + colB0)*K + kk + kselB);
      bf_.u[0] = bp[0]; bf_.u[1] = bp[2];
      acc[0][t] = __builtin_amdgcn_wmma_f32_16x16x32_bf16(false, a0.v, false, bf_.v,
                                                          (short)0, acc[0][t], false, false);
      acc[1][t] = __builtin_amdgcn_wmma_f32_16x16x32_bf16(false, a1.v, false, bf_.v,
                                                          (short)0, acc[1][t], false, false);
    }
  }

  // C/D layout: VGPR r holds row base + r (+8 for lanes 16-31), col = lane&15
  const int c0 = lane & 15;
  const int rsel = (lane >> 4) ? 8 : 0;
  #pragma unroll
  for (int m = 0; m < 2; ++m){
    const int rbase = tileM + m*16 + rsel;
    #pragma unroll
    for (int t = 0; t < 4; ++t){
      int col = tileN + t*16 + c0;
      float bb = bias ? bias[col] : 0.f;
      #pragma unroll
      for (int r = 0; r < 8; ++r){
        C[(size_t)(rbase + r)*Nc + col] = acc[m][t][r] + bb;
      }
    }
  }
}

// ---------------- scoring / sort / output ----------------

__global__ void dots_k(const float* __restrict__ x, const float* __restrict__ linW,
                       float* __restrict__ pr, float* __restrict__ pc){
  int i = blockIdx.x*256 + threadIdx.x;              // NN threads
  const float* row = x + (size_t)i*H;
  float a = 0.f, b = 0.f;
  #pragma unroll 8
  for (int j = 0; j < H; ++j){
    float v = row[j];
    a += v * linW[j];
    b += v * linW[H + j];
  }
  pr[i] = a; pc[i] = b;
}

__global__ void epred_k(const float* __restrict__ pr, const float* __restrict__ pc,
                        const int* __restrict__ ei, const float* __restrict__ linb,
                        float* __restrict__ pred){
  int e = blockIdx.x*256 + threadIdx.x;              // NE threads
  pred[e] = pr[ei[e]] + pc[ei[NE + e]] + linb[0];
}

// per-graph descending argsort of 512 scores (stable via index tiebreak),
// then emit causal / conf edge subsets directly into the output buffer.
__global__ __launch_bounds__(512) void sort_emit_k(const float* __restrict__ pred,
                                                   const int* __restrict__ ei,
                                                   const int* __restrict__ ea,
                                                   float* __restrict__ out){
  __shared__ float sv[EPG];
  __shared__ int   si[EPG];
  const int g = blockIdx.x, t = threadIdx.x;
  const int e0 = g * EPG;
  sv[t] = pred[e0 + t];
  si[t] = t;
  __syncthreads();

  for (int k = 2; k <= EPG; k <<= 1){
    for (int j = k >> 1; j > 0; j >>= 1){
      int ixj = t ^ j;
      if (ixj > t){
        bool up = ((t & k) == 0);
        float v1 = sv[t], v2 = sv[ixj];
        int   i1 = si[t], i2 = si[ixj];
        // "before" in final order = larger value first, ties by smaller index
        bool before2 = (v2 > v1) || (v2 == v1 && i2 < i1);
        if (before2 == up){
          sv[t] = v2; si[t] = i2;
          sv[ixj] = v1; si[ixj] = i1;
        }
      }
      __syncthreads();
    }
  }

  const int e = e0 + si[t];
  if (t < NRES){
    int pos = g*NRES + t;
    out[O_EIC + pos]      = (float)ei[e];
    out[O_EIC + GC + pos] = (float)ei[NE + e];
    out[O_EAC + pos]      = (float)ea[e];
    out[O_PRC + pos]      = sv[t];
  } else {
    int pos = g*NRES + (t - NRES);
    out[O_EIF + pos]      = (float)ei[e];
    out[O_EIF + GC + pos] = (float)ei[NE + e];
    out[O_EAF + pos]      = (float)ea[e];
    out[O_PRF + pos]      = -sv[t];
  }
}

__global__ void finalize_k(const float* __restrict__ x, const int* __restrict__ batch,
                           const float* __restrict__ pred, float* __restrict__ out){
  size_t i = (size_t)blockIdx.x*256 + threadIdx.x;   // NH threads
  float v = x[i];
  out[O_X0 + i] = v;
  out[O_X1 + i] = v;
  if (i < (size_t)NN){
    float bv = (float)batch[i];
    out[O_B0 + i] = bv;
    out[O_B1 + i] = bv;
  }
  if (i < (size_t)NE) out[O_PRED + i] = pred[i];
}

// ---------------- launch ----------------

extern "C" void kernel_launch(void* const* d_in, const int* in_sizes, int n_in,
                              void* d_out, int out_size, void* d_ws, size_t ws_size,
                              hipStream_t stream)
{
  // input map (top-level insertion order; params leaves jax-tree sorted)
  const int*   x_feat   = (const int*)  d_in[0];
  const int*   ei       = (const int*)  d_in[1];   // (2, E) row-major
  const int*   ea       = (const int*)  d_in[2];
  const int*   batch    = (const int*)  d_in[3];
  const float* atom_emb = (const float*)d_in[4];
  // layers[0]: W1,W2,b1,b2,bn1_b,bn1_g,bn_out_b,bn_out_g,bond_emb,eps
  const float* l0_W1    = (const float*)d_in[5];
  const float* l0_W2    = (const float*)d_in[6];
  const float* l0_b1    = (const float*)d_in[7];
  const float* l0_b2    = (const float*)d_in[8];
  const float* l0_bn1b  = (const float*)d_in[9];
  const float* l0_bn1g  = (const float*)d_in[10];
  const float* l0_bnob  = (const float*)d_in[11];
  const float* l0_bnog  = (const float*)d_in[12];
  const float* l0_bond  = (const float*)d_in[13];
  const float* l0_eps   = (const float*)d_in[14];
  const float* l1_W1    = (const float*)d_in[15];
  const float* l1_W2    = (const float*)d_in[16];
  const float* l1_b1    = (const float*)d_in[17];
  const float* l1_b2    = (const float*)d_in[18];
  const float* l1_bn1b  = (const float*)d_in[19];
  const float* l1_bn1g  = (const float*)d_in[20];
  const float* l1_bnob  = (const float*)d_in[21];
  const float* l1_bnog  = (const float*)d_in[22];
  const float* l1_bond  = (const float*)d_in[23];
  const float* l1_eps   = (const float*)d_in[24];
  const float* lin_W    = (const float*)d_in[25];
  const float* lin_b    = (const float*)d_in[26];
  // vn: W1,W2,b1,b2,bn1_b,bn1_g,bn2_b,bn2_g
  const float* vn_W1    = (const float*)d_in[27];
  const float* vn_W2    = (const float*)d_in[28];
  const float* vn_b1    = (const float*)d_in[29];
  const float* vn_b2    = (const float*)d_in[30];
  const float* vn_bn1b  = (const float*)d_in[31];
  const float* vn_bn1g  = (const float*)d_in[32];
  const float* vn_bn2b  = (const float*)d_in[33];
  const float* vn_bn2g  = (const float*)d_in[34];
  const float* vn_emb   = (const float*)d_in[35];

  float* out = (float*)d_out;

  // workspace carve-out
  char* ws = (char*)d_ws;
  size_t off = 0;
  auto alloc = [&](size_t bytes) -> void* {
    void* p = ws + off;
    off += (bytes + 255) & ~(size_t)255;
    return p;
  };
  float*    hinA  = (float*)   alloc(NH  * 4);   // hin0 -> later hin1
  float*    hB    = (float*)   alloc(NH  * 4);   // gemm2 out / h / x
  float*    agg   = (float*)   alloc(NH  * 4);
  uint16_t* zb    = (uint16_t*)alloc(NH  * 2);
  float*    z1    = (float*)   alloc(NH2 * 4);
  uint16_t* z1b   = (uint16_t*)alloc(NH2 * 2);
  uint16_t* W1t0  = (uint16_t*)alloc((size_t)H *H2*2);
  uint16_t* W2t0  = (uint16_t*)alloc((size_t)H2*H *2);
  uint16_t* W1t1  = (uint16_t*)alloc((size_t)H *H2*2);
  uint16_t* W2t1  = (uint16_t*)alloc((size_t)H2*H *2);
  uint16_t* vW1t  = (uint16_t*)alloc((size_t)H *H2*2);
  uint16_t* vW2t  = (uint16_t*)alloc((size_t)H2*H *2);
  float*    mean  = (float*)   alloc(H2 * 4);
  float*    rstd  = (float*)   alloc(H2 * 4);
  float*    vt0   = (float*)   alloc((size_t)G*H *4);
  uint16_t* vt0b  = (uint16_t*)alloc((size_t)G*H *2);
  float*    vt1   = (float*)   alloc((size_t)G*H2*4);
  uint16_t* vt1b  = (uint16_t*)alloc((size_t)G*H2*2);
  float*    vt2   = (float*)   alloc((size_t)G*H *4);
  float*    vnrow = (float*)   alloc((size_t)G*H *4);
  float*    pr    = (float*)   alloc((size_t)NN*4);
  float*    pc    = (float*)   alloc((size_t)NN*4);
  float*    pred  = (float*)   alloc((size_t)NE*4);

  const int TB = 256;
  const int gNH   = (int)(NH  / TB);
  const int gNH2  = (int)(NH2 / TB);
  const int gEmsg = (int)((size_t)NE*(H/4) / TB);
  const int gW    = (H*H2) / TB;

  // --- weight convert + transpose (bf16) ---
  wt_k<<<gW, TB, 0, stream>>>(l0_W1, W1t0, H,  H2);
  wt_k<<<gW, TB, 0, stream>>>(l0_W2, W2t0, H2, H );
  wt_k<<<gW, TB, 0, stream>>>(l1_W1, W1t1, H,  H2);
  wt_k<<<gW, TB, 0, stream>>>(l1_W2, W2t1, H2, H );
  wt_k<<<gW, TB, 0, stream>>>(vn_W1, vW1t, H,  H2);
  wt_k<<<gW, TB, 0, stream>>>(vn_W2, vW2t, H2, H );

  // --- hin0 = atom_emb[x_feat] + vn_emb ---
  embed_k<<<gNH, TB, 0, stream>>>(x_feat, atom_emb, vn_emb, hinA);

  // ================= layer 0 GIN =================
  hipMemsetAsync(agg, 0, NH*4, stream);
  edge_msg_k<<<gEmsg, TB, 0, stream>>>(hinA, l0_bond, ei, ea, agg);
  combine_k <<<gNH,  TB, 0, stream>>>(hinA, agg, l0_eps, zb);
  gemm_bf16_wmma_k<<<dim3(NN/64, H2/256), TB, 0, stream>>>(zb, W1t0, l0_b1, z1, NN, H2, H);
  bn_stats_k<<<H2, TB, 0, stream>>>(z1, NN, H2, mean, rstd);
  bn_apply_k<<<gNH2, TB, 0, stream>>>(z1, mean, rstd, l0_bn1g, l0_bn1b, nullptr, z1, H2, 1);
  tobf_k<<<gNH2, TB, 0, stream>>>(z1, z1b);
  gemm_bf16_wmma_k<<<dim3(NN/64, H/256), TB, 0, stream>>>(z1b, W2t0, l0_b2, hB, NN, H, H2);
  bn_stats_k<<<H, TB, 0, stream>>>(hB, NN, H, mean, rstd);
  // h = relu(bn(h)) + hin0
  bn_apply_k<<<gNH, TB, 0, stream>>>(hB, mean, rstd, l0_bnog, l0_bnob, hinA, hB, H, 1);

  // ================= virtual node =================
  segsum_vn_k<<<(G*H)/TB, TB, 0, stream>>>(hinA, vn_emb, vt0);    // uses hin0
  tobf_k<<<(G*H)/TB, TB, 0, stream>>>(vt0, vt0b);
  gemm_bf16_wmma_k<<<dim3(G/64, H2/256), TB, 0, stream>>>(vt0b, vW1t, vn_b1, vt1, G, H2, H);
  bn_stats_k<<<H2, TB, 0, stream>>>(vt1, G, H2, mean, rstd);
  bn_apply_k<<<(G*H2)/TB, TB, 0, stream>>>(vt1, mean, rstd, vn_bn1g, vn_bn1b, nullptr, vt1, H2, 1);
  tobf_k<<<(G*H2)/TB, TB, 0, stream>>>(vt1, vt1b);
  gemm_bf16_wmma_k<<<dim3(G/64, H/256), TB, 0, stream>>>(vt1b, vW2t, vn_b2, vt2, G, H, H2);
  bn_stats_k<<<H, TB, 0, stream>>>(vt2, G, H, mean, rstd);
  bn_apply_k<<<(G*H)/TB, TB, 0, stream>>>(vt2, mean, rstd, vn_bn2g, vn_bn2b, nullptr, vnrow, H, 1);

  // hin1 = h + vn[batch]  (overwrite hinA)
  add_vn_k<<<gNH, TB, 0, stream>>>(hB, vnrow, hinA);

  // ================= layer 1 GIN =================
  hipMemsetAsync(agg, 0, NH*4, stream);
  edge_msg_k<<<gEmsg, TB, 0, stream>>>(hinA, l1_bond, ei, ea, agg);
  combine_k <<<gNH,  TB, 0, stream>>>(hinA, agg, l1_eps, zb);
  gemm_bf16_wmma_k<<<dim3(NN/64, H2/256), TB, 0, stream>>>(zb, W1t1, l1_b1, z1, NN, H2, H);
  bn_stats_k<<<H2, TB, 0, stream>>>(z1, NN, H2, mean, rstd);
  bn_apply_k<<<gNH2, TB, 0, stream>>>(z1, mean, rstd, l1_bn1g, l1_bn1b, nullptr, z1, H2, 1);
  tobf_k<<<gNH2, TB, 0, stream>>>(z1, z1b);
  gemm_bf16_wmma_k<<<dim3(NN/64, H/256), TB, 0, stream>>>(z1b, W2t1, l1_b2, hB, NN, H, H2);
  bn_stats_k<<<H, TB, 0, stream>>>(hB, NN, H, mean, rstd);
  // x = bn(h) + hin1   (no relu)
  bn_apply_k<<<gNH, TB, 0, stream>>>(hB, mean, rstd, l1_bnog, l1_bnob, hinA, hB, H, 0);

  // ================= scoring + per-graph top-k =================
  dots_k <<<NN/TB, TB, 0, stream>>>(hB, lin_W, pr, pc);
  epred_k<<<NE/TB, TB, 0, stream>>>(pr, pc, ei, lin_b, pred);
  sort_emit_k<<<G, EPG, 0, stream>>>(pred, ei, ea, out);
  finalize_k<<<gNH, TB, 0, stream>>>(hB, batch, pred, out);

  (void)in_sizes; (void)n_in; (void)out_size; (void)ws_size;
}
